// Attentive_12567074308262
// MI455X (gfx1250) — matrix operand order repaired and checked by reading, not compiled, not run on hardware
//
#include <hip/hip_runtime.h>
#include <hip/hip_bf16.h>
#include <math.h>
#include <stdint.h>

// Shapes (fixed by the reference)
#define BB   32
#define TT   32
#define KK   196   // attention locations
#define HH   512   // hidden = A = 512

#define ZT_ROWS   8           // cv rows per TDM chunk
#define ZT_CHUNKS ((KK + ZT_ROWS - 1) / ZT_ROWS)   // 25 (last chunk = 4 rows)

typedef __attribute__((ext_vector_type(16))) __bf16 v16bf;
typedef __attribute__((ext_vector_type(8)))  float  v8f;
typedef __attribute__((ext_vector_type(4)))  unsigned int u32x4;
typedef __attribute__((ext_vector_type(8)))  int i32x8;
typedef __attribute__((ext_vector_type(4)))  int i32x4;

// ---------------------------------------------------------------------------
// Tensor Data Mover: DMA a (nrows x 512) fp32 tile (row stride 512) from
// global into LDS. D# built per CDNA5 ISA sec. 8.3/8.4; 2D tensor, groups
// 2/3 zero. Issued once per wave; completion tracked with TENSORcnt.
// ---------------------------------------------------------------------------
__device__ __forceinline__ void tdm_load_rows_f32(const float* src,
                                                  uint32_t lds_byte_off,
                                                  uint32_t nrows)
{
    const uint64_t ga = (uint64_t)(uintptr_t)src;
    u32x4 g0;
    g0[0] = 1u;                                   // count=1, user descriptor
    g0[1] = lds_byte_off;                         // lds_addr
    g0[2] = (uint32_t)ga;                         // global_addr[31:0]
    g0[3] = (uint32_t)(ga >> 32) | (2u << 30);    // global_addr[56:32] | type=2
    i32x8 g1;
    g1[0] = (int)(2u << 16);                      // wg_mask=0, data_size=4B
    g1[1] = (int)(512u << 16);                    // tensor_dim0[15:0]=512
    g1[2] = (int)(nrows << 16);                   // tensor_dim0 hi=0 | tensor_dim1 lo=nrows
    g1[3] = (int)(512u << 16);                    // tensor_dim1 hi=0 | tile_dim0=512
    g1[4] = (int)nrows;                           // tile_dim1=nrows, tile_dim2=0
    g1[5] = 512;                                  // tensor_dim0_stride lo32
    g1[6] = 0;                                    // stride0 hi | stride1 lo
    g1[7] = 0;                                    // stride1 hi
    i32x4 gz = {0, 0, 0, 0};
#if __clang_major__ >= 23
    i32x8 gz8 = {0, 0, 0, 0, 0, 0, 0, 0};
    __builtin_amdgcn_tensor_load_to_lds(g0, g1, gz, gz, gz8, 0);
#else
    __builtin_amdgcn_tensor_load_to_lds(g0, g1, gz, gz, 0);
#endif
}

// ---------------------------------------------------------------------------
// Pack one 512x512 fp32 weight (row major, KxN) into bf16 WMMA B-fragment
// order: for (n-tile nt, k-step ks, lane L) the 16 bf16 elements a lane
// feeds to v_wmma are contiguous (32B) -> single coalesced load in the GEMM.
// grid = (32 n-tiles, 16 k-steps), block = 32 (one wave).
// ---------------------------------------------------------------------------
__global__ __launch_bounds__(32)
void pack_w_bf16(const float* __restrict__ W, __bf16* __restrict__ Wp)
{
    const int nt   = blockIdx.x;
    const int ks   = blockIdx.y;
    const int lane = threadIdx.x & 31;
    const int half = lane >> 4;
    const int n    = nt * 16 + (lane & 15);
    const int k0   = ks * 32;
    v16bf frag;
    #pragma unroll
    for (int j = 0; j < 16; ++j)        // B layout: k = j + (j>=8 ? 8 : 0) + half*8
        frag[j] = (__bf16)W[(k0 + j + (j >= 8 ? 8 : 0) + half * 8) * HH + n];
    ((v16bf*)Wp)[(nt * 16 + ks) * 32 + lane] = frag;
}

// ---------------------------------------------------------------------------
// C[M x 512] = A[M x 512] @ W[512 x 512] (+ addC), W pre-packed bf16.
// One wave per 16x16 C tile; 16 k-steps of v_wmma_f32_16x16x32_bf16.
// ---------------------------------------------------------------------------
template <bool HAS_ADD>
__global__ __launch_bounds__(32)
void gemm512_bf16wmma(const float* __restrict__ Am,
                      const __bf16* __restrict__ Wp,
                      const float* __restrict__ addC,
                      float* __restrict__ Cm)
{
    const int lane = threadIdx.x & 31;
    const int half = lane >> 4;
    const int m0   = blockIdx.x * 16;
    const int n0   = blockIdx.y * 16;
    const int mrow = m0 + (lane & 15);
    const int ncol = n0 + (lane & 15);

    const float* arow  = Am + (size_t)mrow * HH;
    const v16bf* WpV   = (const v16bf*)Wp + (size_t)blockIdx.y * 16 * 32 + lane;

    v8f acc = {};
    #pragma unroll 4
    for (int ks = 0; ks < 16; ++ks) {
        const int k0 = ks * 32;
        // A (16x32): element j -> k = j + (j>=8 ? 8 : 0) + half*8
        const float4 a0 = *(const float4*)(arow + k0 + half * 8);
        const float4 a1 = *(const float4*)(arow + k0 + half * 8 + 4);
        const float4 a2 = *(const float4*)(arow + k0 + 16 + half * 8);
        const float4 a3 = *(const float4*)(arow + k0 + 16 + half * 8 + 4);
        v16bf af;
        af[0]  = (__bf16)a0.x; af[1]  = (__bf16)a0.y; af[2]  = (__bf16)a0.z; af[3]  = (__bf16)a0.w;
        af[4]  = (__bf16)a1.x; af[5]  = (__bf16)a1.y; af[6]  = (__bf16)a1.z; af[7]  = (__bf16)a1.w;
        af[8]  = (__bf16)a2.x; af[9]  = (__bf16)a2.y; af[10] = (__bf16)a2.z; af[11] = (__bf16)a2.w;
        af[12] = (__bf16)a3.x; af[13] = (__bf16)a3.y; af[14] = (__bf16)a3.z; af[15] = (__bf16)a3.w;

        const v16bf bf = WpV[ks * 32];  // one 32B coalesced load

        acc = __builtin_amdgcn_wmma_f32_16x16x32_bf16(
            false, af, false, bf, (short)0, acc, false, false);
    }
    #pragma unroll
    for (int r = 0; r < 8; ++r) {
        const int idx = (m0 + r + half * 8) * HH + ncol;
        if (HAS_ADD) Cm[idx] = acc[r] + addC[idx];
        else         Cm[idx] = acc[r];
    }
}

// ---------------------------------------------------------------------------
// z_t[b,t,k] = sum_a tanh(cv[b,k,a] + cg[b,t,a]) * wh[a]
// z_ext[b,t] = sum_a tanh(cs[b,t,a]) * wh[a]
// Block per (b,t). cg/wh staged in LDS; cv streamed through LDS in 8-row
// chunks by the Tensor Data Mover, double-buffered: TDM(c+1) overlaps the
// tanh/reduce of chunk c; s_wait_tensorcnt + barrier close each stage.
// ---------------------------------------------------------------------------
__global__ __launch_bounds__(256)
void zt_kernel(const float* __restrict__ cv,   // (B*K, 512)
               const float* __restrict__ cg,   // (B*T, 512)
               const float* __restrict__ cs,   // (B*T, 512)
               const float* __restrict__ wh,   // (512)
               float* __restrict__ z,          // (B*T, 196)
               float* __restrict__ zext)       // (B*T)
{
    __shared__ float scg[HH];
    __shared__ float swh[HH];
    __shared__ float buf[2][ZT_ROWS * HH];     // 2 x 16 KB

    const int bt  = blockIdx.x;
    const int b   = bt >> 5;
    const int tid = threadIdx.x;
    const int lane = tid & 31;
    const int wave = tid >> 5;

    for (int i = tid; i < HH; i += 256) {
        scg[i] = cg[bt * HH + i];
        swh[i] = wh[i];
    }

    const float* cvb = cv + (size_t)b * KK * HH;

    // prologue: chunk 0 into buf[0]
    if (tid < 32) {
        tdm_load_rows_f32(cvb, (uint32_t)(uintptr_t)&buf[0][0], ZT_ROWS);
        __builtin_amdgcn_s_wait_tensorcnt(0);
    }
    __syncthreads();

    for (int c = 0; c < ZT_CHUNKS; ++c) {
        if (tid < 32 && (c + 1) < ZT_CHUNKS) {
            const int rem = KK - (c + 1) * ZT_ROWS;
            const uint32_t nr = (rem < ZT_ROWS) ? (uint32_t)rem : (uint32_t)ZT_ROWS;
            tdm_load_rows_f32(cvb + (size_t)(c + 1) * ZT_ROWS * HH,
                              (uint32_t)(uintptr_t)&buf[(c + 1) & 1][0], nr);
        }
        // each wave reduces one row of the current chunk
        const int k = c * ZT_ROWS + wave;
        if (k < KK) {
            const float* row = &buf[c & 1][wave * HH];
            float acc = 0.f;
            #pragma unroll
            for (int i = 0; i < 16; ++i) {
                const int a = lane + i * 32;
                acc = fmaf(tanhf(row[a] + scg[a]), swh[a], acc);
            }
            #pragma unroll
            for (int off = 16; off > 0; off >>= 1)
                acc += __shfl_xor(acc, off, 32);
            if (lane == 0) z[(size_t)bt * KK + k] = acc;
        }
        if (tid < 32) __builtin_amdgcn_s_wait_tensorcnt(0);
        __syncthreads();
    }

    // sentinel branch: z_ext = sum_a tanh(cs[bt,a]) * wh[a]  (wave 0)
    if (tid < 32) {
        const float* srow = cs + (size_t)bt * HH;
        float acc = 0.f;
        #pragma unroll
        for (int i = 0; i < 16; ++i) {
            const int a = lane + i * 32;
            acc = fmaf(tanhf(srow[a]), swh[a], acc);
        }
        #pragma unroll
        for (int off = 16; off > 0; off >>= 1)
            acc += __shfl_xor(acc, off, 32);
        if (lane == 0) zext[bt] = acc;
    }
}

// ---------------------------------------------------------------------------
// Per (b,t): alpha = softmax(z); extended 197-way softmax gives beta;
// c_t = alpha @ att_feats[b]; c_hat = beta*sent + (1-beta)*c_t.
// ---------------------------------------------------------------------------
__global__ __launch_bounds__(256)
void softmax_ct_kernel(const float* __restrict__ z,
                       const float* __restrict__ zext,
                       const float* __restrict__ att,
                       const float* __restrict__ sent,
                       float* __restrict__ out_chat,
                       float* __restrict__ out_alpha,
                       float* __restrict__ out_beta)
{
    __shared__ float s_alpha[KK];
    __shared__ float red[256];
    __shared__ float s_beta;
    const int bt  = blockIdx.x;
    const int b   = bt >> 5;
    const int tid = threadIdx.x;

    const float zi = (tid < KK) ? z[(size_t)bt * KK + tid] : -INFINITY;

    red[tid] = zi;
    __syncthreads();
    #pragma unroll
    for (int s = 128; s > 0; s >>= 1) {
        if (tid < s) red[tid] = fmaxf(red[tid], red[tid + s]);
        __syncthreads();
    }
    const float m196 = red[0];
    __syncthreads();

    const float e = (tid < KK) ? __expf(zi - m196) : 0.f;
    red[tid] = e;
    __syncthreads();
    #pragma unroll
    for (int s = 128; s > 0; s >>= 1) {
        if (tid < s) red[tid] += red[tid + s];
        __syncthreads();
    }
    const float sum196 = red[0];
    __syncthreads();

    const float alpha = e / sum196;
    if (tid < KK) {
        s_alpha[tid] = alpha;
        out_alpha[(size_t)bt * KK + tid] = alpha;
    }

    const float ze     = zext[bt];
    const float m197   = fmaxf(m196, ze);
    const float sum197 = sum196 * __expf(m196 - m197) + __expf(ze - m197);
    const float beta   = __expf(ze - m197) / sum197;
    if (tid == 0) { out_beta[bt] = beta; s_beta = beta; }
    __syncthreads();

    const float bta = s_beta;
    for (int h = tid; h < HH; h += 256) {
        float acc = 0.f;
        for (int k = 0; k < KK; ++k)
            acc = fmaf(s_alpha[k], att[(size_t)(b * KK + k) * HH + h], acc);
        out_chat[(size_t)bt * HH + h] =
            bta * sent[(size_t)bt * HH + h] + (1.f - bta) * acc;
    }
}

// ---------------------------------------------------------------------------
extern "C" void kernel_launch(void* const* d_in, const int* in_sizes, int n_in,
                              void* d_out, int out_size, void* d_ws, size_t ws_size,
                              hipStream_t stream)
{
    (void)in_sizes; (void)n_in; (void)out_size; (void)ws_size;
    const float* att = (const float*)d_in[0];  // (32,196,512)
    const float* hid = (const float*)d_in[1];  // (32, 32,512)
    const float* sen = (const float*)d_in[2];  // (32, 32,512)
    const float* Wv  = (const float*)d_in[3];  // (512,512)
    const float* Wg  = (const float*)d_in[4];  // (512,512)
    const float* Ws  = (const float*)d_in[5];  // (512,512)
    const float* wh  = (const float*)d_in[6];  // (512)

    // Workspace: [bf16 packed weights x3][cv][cg][cs][z][zext]
    __bf16* wvp = (__bf16*)d_ws;                 // 512*512 bf16, 32B aligned
    __bf16* wgp = wvp + (size_t)HH * HH;
    __bf16* wsp = wgp + (size_t)HH * HH;
    float*  fws = (float*)(wsp + (size_t)HH * HH);
    float* cv = fws;                             // 6272*512
    float* cg = cv + (size_t)BB * KK * HH;       // 1024*512
    float* cs = cg + (size_t)BB * TT * HH;       // 1024*512
    float* zz = cs + (size_t)BB * TT * HH;       // 1024*196
    float* ze = zz + (size_t)BB * TT * KK;       // 1024

    float* out_chat  = (float*)d_out;
    float* out_alpha = out_chat + (size_t)BB * TT * HH;
    float* out_beta  = out_alpha + (size_t)BB * TT * KK;

    // pack weights to bf16 fragment order (tiny, L2-resident afterwards)
    pack_w_bf16<<<dim3(HH / 16, HH / 32), dim3(32), 0, stream>>>(Wv, wvp);
    pack_w_bf16<<<dim3(HH / 16, HH / 32), dim3(32), 0, stream>>>(Wg, wgp);
    pack_w_bf16<<<dim3(HH / 16, HH / 32), dim3(32), 0, stream>>>(Ws, wsp);

    // cv = att @ Wv ; cg = hid @ Wg ; cs = sen @ Ws + cg
    gemm512_bf16wmma<false><<<dim3((BB * KK) / 16, HH / 16), dim3(32), 0, stream>>>(
        att, wvp, (const float*)nullptr, cv);
    gemm512_bf16wmma<false><<<dim3((BB * TT) / 16, HH / 16), dim3(32), 0, stream>>>(
        hid, wgp, (const float*)nullptr, cg);
    gemm512_bf16wmma<true><<<dim3((BB * TT) / 16, HH / 16), dim3(32), 0, stream>>>(
        sen, wsp, cg, cs);

    // fused tanh/dot pass, cv streamed via Tensor Data Mover
    zt_kernel<<<dim3(BB * TT), dim3(256), 0, stream>>>(cv, cg, cs, wh, zz, ze);

    // softmaxes, c_t, sentinel blend
    softmax_ct_kernel<<<dim3(BB * TT), dim3(256), 0, stream>>>(
        zz, ze, att, sen, out_chat, out_alpha, out_beta);
}
